// Model_88055419502655
// MI455X (gfx1250) — matrix operand order repaired and checked by reading, not compiled, not run on hardware
//
#include <hip/hip_runtime.h>
#include <math.h>

#define DM 512
#define NH 8
#define DH 64
#define NEG_BIG (-1e30f)
#define SCALE_ATTN 0.125f                   /* 1/sqrt(64) */
#define BN_SCALE_C 0.9999950000374997f      /* 1/sqrt(1+1e-5) */
#define UMAX 48
#define KSTEP 64

typedef __attribute__((ext_vector_type(16))) __bf16       v16bf;
typedef __attribute__((ext_vector_type(8)))  float        v8f;
typedef __attribute__((ext_vector_type(4)))  unsigned int u32x4;

union FragBF { v16bf v; u32x4 q[2]; };

__device__ __forceinline__ unsigned short f2bf(float f) {
  unsigned int u = __float_as_uint(f);
  unsigned int r = u + 0x7FFFu + ((u >> 16) & 1u);
  return (unsigned short)(r >> 16);
}
/* pack two f32 -> two bf16 in one dword, pure 32-bit ops (no b16 moves) */
__device__ __forceinline__ unsigned int pack2bf(float x, float y) {
  unsigned int ux = __float_as_uint(x);
  unsigned int uy = __float_as_uint(y);
  ux = ux + 0x7FFFu + ((ux >> 16) & 1u);
  uy = uy + 0x7FFFu + ((uy >> 16) & 1u);
  return (ux >> 16) | (uy & 0xFFFF0000u);
}
__device__ __forceinline__ float gelu_f(float x) {
  float x3 = x * x * x;
  return 0.5f * x * (1.0f + tanhf(0.7978845608028654f * (x + 0.044715f * x3)));
}

/* ------------------------------------------------------------------ */
/* WMMA GEMM: C[M,N] = act(rowmap(A)[M,K] @ W + bias) (+Res) (+=C)     */
/* Requires M%64==0, N%64==0, K%64==0 (guaranteed by host).            */
/* logical row m -> physical row (m/Lb)*batchRows + (m%Lb);            */
/* batchRows==0 means identity mapping.                                */
/* W element [k][n] at W[k*wRS + n*wCS]                                */
/* ------------------------------------------------------------------ */
__global__ __launch_bounds__(128) void k_gemm(
    const float* __restrict__ A, int lda, int Lb, int batchRows,
    const float* __restrict__ W, int wRS, int wCS,
    const float* __restrict__ bias, const float* __restrict__ Res,
    float* __restrict__ C, int M, int N, int K, int act, int accum)
{
  __shared__ __align__(16) unsigned int   sA32[64 * KSTEP / 2]; /* [m][k] bf16x2 */
  __shared__ __align__(16) unsigned short sB[64 * KSTEP];       /* [n][k] bf16   */

  const int tid  = threadIdx.x;
  const int lane = tid & 31;
  const int wave = tid >> 5;
  const int hf   = lane >> 4;      /* half-wave: 0 or 1 */
  const int lr   = lane & 15;
  const int m0   = blockIdx.y * 64;
  const int n0   = blockIdx.x * 64;
  const int wm   = (wave >> 1) * 32;
  const int wn   = (wave & 1) * 32;

  v8f acc[2][2];
#pragma unroll
  for (int mi = 0; mi < 2; ++mi)
#pragma unroll
    for (int ni = 0; ni < 2; ++ni)
#pragma unroll
      for (int e = 0; e < 8; ++e) acc[mi][ni][e] = 0.0f;

  /* ---- hoist per-thread A source pointers (row remap idiv out of k-loop) ---- */
  const float* aptr[8];
  int aoff[8];   /* dword index into sA32 */
#pragma unroll
  for (int j = 0; j < 8; ++j) {
    int i = tid + j * 128;
    int r = i >> 4, c = (i & 15) << 2;
    int m = m0 + r;
    long pr;
    if (batchRows == 0) pr = m;
    else pr = (long)(m / Lb) * batchRows + (m % Lb);
    aptr[j] = A + pr * (long)lda + c;
    aoff[j] = r * (KSTEP / 2) + (c >> 1);
  }
  /* ---- hoist per-thread W source pointers for the wCS==1 fast path ---- */
  const float* wptr[8];
  int wkk[8], wnn[8];
#pragma unroll
  for (int j = 0; j < 8; ++j) {
    int i = tid + j * 128;
    int kk = i >> 4, n = (i & 15) << 2;
    wptr[j] = W + (long)kk * wRS + n0 + n;
    wkk[j] = kk; wnn[j] = n;
  }

  for (int k0 = 0; k0 < K; k0 += KSTEP) {
    /* batch all A tile loads (8 x b128) before any conversion */
    float4 va[8];
#pragma unroll
    for (int j = 0; j < 8; ++j) va[j] = *(const float4*)(aptr[j] + k0);

    if (wCS == 1) {
      /* batch W tile loads too, then convert/store everything */
      float4 vb[8];
      const long wk0 = (long)k0 * wRS;
#pragma unroll
      for (int j = 0; j < 8; ++j) vb[j] = *(const float4*)(wptr[j] + wk0);
#pragma unroll
      for (int j = 0; j < 8; ++j) {
        uint2 pk;
        pk.x = pack2bf(va[j].x, va[j].y);
        pk.y = pack2bf(va[j].z, va[j].w);
        *(uint2*)(&sA32[aoff[j]]) = pk;
      }
#pragma unroll
      for (int j = 0; j < 8; ++j) {
        int kk = wkk[j], n = wnn[j];
        sB[(n + 0) * KSTEP + kk] = f2bf(vb[j].x);
        sB[(n + 1) * KSTEP + kk] = f2bf(vb[j].y);
        sB[(n + 2) * KSTEP + kk] = f2bf(vb[j].z);
        sB[(n + 3) * KSTEP + kk] = f2bf(vb[j].w);
      }
    } else {
#pragma unroll
      for (int j = 0; j < 8; ++j) {
        uint2 pk;
        pk.x = pack2bf(va[j].x, va[j].y);
        pk.y = pack2bf(va[j].z, va[j].w);
        *(uint2*)(&sA32[aoff[j]]) = pk;
      }
      /* strided weights (conv taps): pair-packed scalar gather */
#pragma unroll
      for (int j = 0; j < 16; ++j) {
        int i = tid + j * 128;        /* 2048 k-pairs */
        int n = i & 63, kp = i >> 6;  /* kp 0..31 */
        float v0 = W[(long)(k0 + 2 * kp)     * wRS + (long)(n0 + n) * wCS];
        float v1 = W[(long)(k0 + 2 * kp + 1) * wRS + (long)(n0 + n) * wCS];
        *(unsigned int*)(&sB[n * KSTEP + 2 * kp]) = pack2bf(v0, v1);
      }
    }
    if (k0 + KSTEP < K)  /* gfx1250: lowers to global_prefetch_b8 */
      __builtin_prefetch(W + (long)(k0 + KSTEP) * wRS + n0, 0, 0);
    __syncthreads();

    const u32x4* pA = (const u32x4*)sA32;  /* 8 bf16 per u32x4 */
    const u32x4* pB = (const u32x4*)sB;
#pragma unroll
    for (int ks = 0; ks < 2; ++ks) {
      FragBF fa[2], fb[2];
#pragma unroll
      for (int mi = 0; mi < 2; ++mi) {
        int row = wm + mi * 16 + lr;
        /* 16-bit A layout: lane half h holds K = 8h..8h+7 and 16+8h..16+8h+7 */
        fa[mi].q[0] = pA[row * 8 + ks * 4 + hf];
        fa[mi].q[1] = pA[row * 8 + ks * 4 + 2 + hf];
      }
#pragma unroll
      for (int ni = 0; ni < 2; ++ni) {
        int col = wn + ni * 16 + lr;
        /* 16-bit B layout: lane half h holds K = 8h..8h+7 and 16+8h..16+8h+7 */
        fb[ni].q[0] = pB[col * 8 + ks * 4 + hf];
        fb[ni].q[1] = pB[col * 8 + ks * 4 + 2 + hf];
      }
#pragma unroll
      for (int mi = 0; mi < 2; ++mi)
#pragma unroll
        for (int ni = 0; ni < 2; ++ni)
          acc[mi][ni] = __builtin_amdgcn_wmma_f32_16x16x32_bf16(
              false, fa[mi].v, false, fb[ni].v, (short)0, acc[mi][ni], false, false);
    }
    __syncthreads();
  }

  /* epilogue: C/D layout -> m = r + 8*half, n = lane%16 (no bounds: M,N %64==0) */
#pragma unroll
  for (int mi = 0; mi < 2; ++mi)
#pragma unroll
    for (int ni = 0; ni < 2; ++ni)
#pragma unroll
      for (int r = 0; r < 8; ++r) {
        int m = m0 + wm + mi * 16 + hf * 8 + r;
        int n = n0 + wn + ni * 16 + lr;
        long ci = (long)m * N + n;
        float v = acc[mi][ni][r];
        if (accum) v += C[ci];
        if (bias)  v += bias[n];
        if (act == 1) v = gelu_f(v);
        if (Res)   v += Res[ci];
        C[ci] = v;
      }
}

/* -------------------- data embedding -------------------- */
__global__ void k_embed(const float* __restrict__ x, const float* __restrict__ xm,
                        const float* __restrict__ cw, const float* __restrict__ tw,
                        const float* __restrict__ tb, float* __restrict__ out,
                        int L, int Cin)
{
  long id = (long)blockIdx.x * blockDim.x + threadIdx.x;
  long total = (long)4 * L * DM;
  if (id >= total) return;
  int d = (int)(id % DM);
  int l = (int)((id / DM) % L);
  int b = (int)(id / ((long)DM * L));
  float acc = 0.0f;
#pragma unroll
  for (int t = 0; t < 3; ++t) {
    int r = l + t - 1;
    if (r < 0) r = L - 1;
    if (r >= L) r = 0;
    const float* xr = x + (long)(b * L + r) * Cin;
    for (int i = 0; i < Cin; ++i)
      acc += xr[i] * cw[(d * Cin + i) * 3 + t];
  }
  int dd = d & ~1;
  float freq = expf((float)dd * (-9.210340371976184f / (float)DM));
  float ang = (float)l * freq;
  acc += (d & 1) ? cosf(ang) : sinf(ang);
  const float* xmr = xm + (long)(b * L + l) * 4;
  float ta = tb[d];
#pragma unroll
  for (int f = 0; f < 4; ++f) ta += xmr[f] * tw[f * DM + d];
  out[id] = acc + ta;
}

/* -------------------- layer norm (row of 512) -------------------- */
__global__ __launch_bounds__(256) void k_layernorm(
    const float* __restrict__ in, const float* __restrict__ g,
    const float* __restrict__ bb, float* __restrict__ out)
{
  __shared__ float red[256];
  long row = blockIdx.x;
  const float* x = in + row * DM;
  int t = threadIdx.x;
  float s = 0.f;
  for (int i = t; i < DM; i += 256) s += x[i];
  red[t] = s; __syncthreads();
  for (int st = 128; st > 0; st >>= 1) { if (t < st) red[t] += red[t + st]; __syncthreads(); }
  float mu = red[0] / (float)DM; __syncthreads();
  float v = 0.f;
  for (int i = t; i < DM; i += 256) { float d = x[i] - mu; v += d * d; }
  red[t] = v; __syncthreads();
  for (int st = 128; st > 0; st >>= 1) { if (t < st) red[t] += red[t + st]; __syncthreads(); }
  float inv = rsqrtf(red[0] / (float)DM + 1e-5f); __syncthreads();
  float* o = out + row * DM;
  for (int i = t; i < DM; i += 256) o[i] = (x[i] - mu) * inv * g[i] + bb[i];
}

/* -------------------- ProbSparse attention pieces -------------------- */
__device__ __forceinline__ int samp_idx(int l, int j, int LK) {
  unsigned h = (unsigned)l * 2654435761u + (unsigned)j * 40503u + 0x9E3779B9u;
  h ^= h >> 13; h *= 0x85EBCA6Bu; h ^= h >> 16;
  return (int)(h % (unsigned)LK);
}

__global__ void k_probM(const float* __restrict__ Q, const float* __restrict__ Kp,
                        float* __restrict__ Mout, int LQ, int LK, int UP)
{
  long id = (long)blockIdx.x * blockDim.x + threadIdx.x;
  if (id >= 32L * LQ) return;
  int l = (int)(id % LQ);
  int bh = (int)(id / LQ);
  int h = bh & 7, b = bh >> 3;
  const float* q = Q + (long)(b * LQ + l) * DM + h * DH;
  float mx = -3.4e38f, sm = 0.f;
  for (int j = 0; j < UP; ++j) {
    int ki = samp_idx(l, j, LK);
    const float* kk = Kp + (long)(b * LK + ki) * DM + h * DH;
    float d = 0.f;
    for (int t = 0; t < DH; ++t) d += q[t] * kk[t];
    mx = fmaxf(mx, d); sm += d;
  }
  Mout[id] = mx - sm / (float)LK;
}

__global__ __launch_bounds__(256) void k_topk(const float* __restrict__ Mv,
                                              int* __restrict__ top, int LQ, int u)
{
  int bh = blockIdx.x;
  __shared__ float sM[1024];
  __shared__ float rv[256];
  __shared__ int   ri[256];
  int t = threadIdx.x;
  for (int i = t; i < LQ; i += 256) sM[i] = Mv[(long)bh * LQ + i];
  __syncthreads();
  for (int it = 0; it < u; ++it) {
    float bv = -3.4e38f; int bi = 0;
    for (int i = t; i < LQ; i += 256)
      if (sM[i] > bv) { bv = sM[i]; bi = i; }
    rv[t] = bv; ri[t] = bi; __syncthreads();
    for (int st = 128; st > 0; st >>= 1) {
      if (t < st && rv[t + st] > rv[t]) { rv[t] = rv[t + st]; ri[t] = ri[t + st]; }
      __syncthreads();
    }
    if (t == 0) { top[bh * UMAX + it] = ri[0]; sM[ri[0]] = -3.4e38f; }
    __syncthreads();
  }
}

__global__ __launch_bounds__(128) void k_scores(
    const float* __restrict__ Q, const float* __restrict__ Kp,
    const int* __restrict__ top, float* __restrict__ S, int LQ, int LK, int u)
{
  int blk = blockIdx.x;
  int ui = blk % u, bh = blk / u;
  int h = bh & 7, b = bh >> 3;
  int qrow = top[bh * UMAX + ui];
  const float* q = Q + (long)(b * LQ + qrow) * DM + h * DH;
  float* srow = S + (long)blk * LK;
  for (int k = threadIdx.x; k < LK; k += 128) {
    const float* kk = Kp + (long)(b * LK + k) * DM + h * DH;
    float d = 0.f;
    for (int t = 0; t < DH; ++t) d += q[t] * kk[t];
    srow[k] = d * SCALE_ATTN;
  }
}

__global__ __launch_bounds__(128) void k_softmax(float* __restrict__ S,
    const int* __restrict__ top, int LK, int u, int masked)
{
  int blk = blockIdx.x;
  int ui = blk % u, bh = blk / u;
  float* row = S + (long)blk * LK;
  int qi = top[bh * UMAX + ui];
  __shared__ float sh[128];
  int t = threadIdx.x;
  float mx = -3.4e38f;
  for (int k = t; k < LK; k += 128) {
    float v = row[k];
    if (masked && k > qi) { v = NEG_BIG; row[k] = v; }
    mx = fmaxf(mx, v);
  }
  sh[t] = mx; __syncthreads();
  for (int st = 64; st > 0; st >>= 1) { if (t < st) sh[t] = fmaxf(sh[t], sh[t + st]); __syncthreads(); }
  mx = sh[0]; __syncthreads();
  float sum = 0.f;
  for (int k = t; k < LK; k += 128) { float e = expf(row[k] - mx); row[k] = e; sum += e; }
  sh[t] = sum; __syncthreads();
  for (int st = 64; st > 0; st >>= 1) { if (t < st) sh[t] += sh[t + st]; __syncthreads(); }
  float inv = 1.0f / sh[0];
  for (int k = t; k < LK; k += 128) row[k] *= inv;
}

__global__ void k_ctx_mean(const float* __restrict__ V, float* __restrict__ ctx,
                           int LQ, int LK)
{
  int id = blockIdx.x * blockDim.x + threadIdx.x;
  if (id >= 2048) return;
  int d = id & 63, h = (id >> 6) & 7, b = id >> 9;
  float s = 0.f;
  for (int l = 0; l < LK; ++l) s += V[(long)(b * LK + l) * DM + h * DH + d];
  s /= (float)LK;
  for (int l = 0; l < LQ; ++l) ctx[(long)(b * LQ + l) * DM + h * DH + d] = s;
}

__global__ void k_ctx_cumsum(const float* __restrict__ V, float* __restrict__ ctx, int L)
{
  int id = blockIdx.x * blockDim.x + threadIdx.x;
  if (id >= 2048) return;
  int d = id & 63, h = (id >> 6) & 7, b = id >> 9;
  float s = 0.f;
  for (int l = 0; l < L; ++l) {
    s += V[(long)(b * L + l) * DM + h * DH + d];
    ctx[(long)(b * L + l) * DM + h * DH + d] = s;
  }
}

__global__ __launch_bounds__(64) void k_scatter(
    const float* __restrict__ S, const float* __restrict__ V,
    const int* __restrict__ top, float* __restrict__ ctx, int LQ, int LK, int u)
{
  int blk = blockIdx.x;
  int ui = blk % u, bh = blk / u;
  int h = bh & 7, b = bh >> 3;
  int d = threadIdx.x;
  const float* row = S + (long)blk * LK;
  float acc = 0.f;
  for (int k = 0; k < LK; ++k)
    acc += row[k] * V[(long)(b * LK + k) * DM + h * DH + d];
  int m = top[bh * UMAX + ui];
  ctx[(long)(b * LQ + m) * DM + h * DH + d] = acc;
}

/* -------------------- conv-layer helpers -------------------- */
__global__ void k_padcirc(const float* __restrict__ in, float* __restrict__ out, int L)
{
  long id = (long)blockIdx.x * blockDim.x + threadIdx.x;
  long total = (long)4 * (L + 2) * DM;
  if (id >= total) return;
  int d = (int)(id % DM);
  long t = id / DM;
  int r = (int)(t % (L + 2));
  int b = (int)(t / (L + 2));
  int sr = (r == 0) ? (L - 1) : ((r == L + 1) ? 0 : r - 1);
  out[id] = in[(long)(b * L + sr) * DM + d];
}

__global__ void k_eluscale(float* __restrict__ x, long n)
{
  long id = (long)blockIdx.x * blockDim.x + threadIdx.x;
  if (id >= n) return;
  float v = x[id] * BN_SCALE_C;
  x[id] = v > 0.f ? v : (expf(v) - 1.f);
}

__global__ void k_maxpool(const float* __restrict__ in, float* __restrict__ out,
                          int L, int Lo)
{
  long id = (long)blockIdx.x * blockDim.x + threadIdx.x;
  long total = (long)4 * Lo * DM;
  if (id >= total) return;
  int d = (int)(id % DM);
  int lo = (int)((id / DM) % Lo);
  int b = (int)(id / ((long)DM * Lo));
  float m = -3.4e38f;
#pragma unroll
  for (int t = 0; t < 3; ++t) {
    int li = 2 * lo - 1 + t;
    if (li >= 0 && li < L) m = fmaxf(m, in[(long)(b * L + li) * DM + d]);
  }
  out[id] = m;
}

/* -------------------- final projection -------------------- */
__global__ void k_proj(const float* __restrict__ x, const float* __restrict__ pw,
                       const float* __restrict__ pb, float* __restrict__ out)
{
  int id = blockIdx.x * blockDim.x + threadIdx.x;
  if (id >= 4 * 512) return;
  int p = id & 511, b = id >> 9;
  const float* row = x + (long)(b * 1024 + 512 + p) * DM;
  float s = pb[0];
  for (int i = 0; i < DM; ++i) s += row[i] * pw[i];
  out[id] = s;
}

/* ================== host side ================== */
struct AttnP { const float *bk, *bo, *bq, *bv, *wk, *wo, *wq, *wv; };
struct EncLayerP {
  AttnP attn;
  const float *ff1_b, *ff1_w, *ff2_b, *ff2_w;
  const float *ln1_b, *ln1_g, *ln2_b, *ln2_g;
};
struct DecLayerP {
  AttnP cross;
  const float *ff1_b, *ff1_w, *ff2_b, *ff2_w;
  const float *ln1_b, *ln1_g, *ln2_b, *ln2_g, *ln3_b, *ln3_g;
  AttnP self;
};

static void gemm(hipStream_t st, const float* A, int lda, int Lb, int batchRows,
                 const float* W, int wRS, int wCS, const float* bias,
                 const float* Res, float* C, int M, int N, int K, int act, int accum)
{
  dim3 g((unsigned)((N + 63) / 64), (unsigned)((M + 63) / 64));
  k_gemm<<<g, 128, 0, st>>>(A, lda, Lb, batchRows, W, wRS, wCS, bias, Res, C, M, N, K, act, accum);
}

static void run_attn(hipStream_t st, const float* Q, const float* K, const float* V,
                     float* ctx, float* Mbuf, float* Sbuf, int* Top,
                     int LQ, int LK, int u, int UP, int masked)
{
  long tM = 32L * LQ;
  k_probM<<<(unsigned)((tM + 127) / 128), 128, 0, st>>>(Q, K, Mbuf, LQ, LK, UP);
  k_topk<<<32, 256, 0, st>>>(Mbuf, Top, LQ, u);
  k_scores<<<32 * u, 128, 0, st>>>(Q, K, Top, Sbuf, LQ, LK, u);
  k_softmax<<<32 * u, 128, 0, st>>>(Sbuf, Top, LK, u, masked);
  if (masked) k_ctx_cumsum<<<16, 128, 0, st>>>(V, ctx, LQ);
  else        k_ctx_mean<<<16, 128, 0, st>>>(V, ctx, LQ, LK);
  k_scatter<<<32 * u, 64, 0, st>>>(Sbuf, V, Top, ctx, LQ, LK, u);
}

static int uc_factor(int L) {
  int v = 5 * (int)ceil(log((double)L));
  return v < L ? v : L;
}

extern "C" void kernel_launch(void* const* d_in, const int* in_sizes, int n_in,
                              void* d_out, int out_size, void* d_ws, size_t ws_size,
                              hipStream_t stream)
{
  (void)in_sizes; (void)n_in; (void)out_size; (void)ws_size;
  auto F = [&](int i) { return (const float*)d_in[i]; };

  const float* x_enc  = F(0);
  const float* xm_enc = F(1);
  const float* x_dec  = F(2);
  const float* xm_dec = F(3);

  int p = 4;
  auto read_attn = [&](void) {
    AttnP a;
    a.bk = F(p++); a.bo = F(p++); a.bq = F(p++); a.bv = F(p++);
    a.wk = F(p++); a.wo = F(p++); a.wq = F(p++); a.wv = F(p++);
    return a;
  };

  /* params flattened alphabetically (jax pytree order) */
  const float *convL_b[2], *convL_w[2];
  for (int i = 0; i < 2; ++i) { convL_b[i] = F(p++); convL_w[i] = F(p++); }
  const float* dec_conv_w = F(p++);
  const float* dec_time_b = F(p++);
  const float* dec_time_w = F(p++);
  DecLayerP dl[2];
  for (int i = 0; i < 2; ++i) {
    dl[i].cross = read_attn();
    dl[i].ff1_b = F(p++); dl[i].ff1_w = F(p++); dl[i].ff2_b = F(p++); dl[i].ff2_w = F(p++);
    dl[i].ln1_b = F(p++); dl[i].ln1_g = F(p++); dl[i].ln2_b = F(p++); dl[i].ln2_g = F(p++);
    dl[i].ln3_b = F(p++); dl[i].ln3_g = F(p++);
    dl[i].self = read_attn();
  }
  const float* dec_norm_b = F(p++);
  const float* dec_norm_g = F(p++);
  const float* enc_conv_w = F(p++);
  const float* enc_time_b = F(p++);
  const float* enc_time_w = F(p++);
  EncLayerP el[3];
  for (int i = 0; i < 3; ++i) {
    el[i].attn = read_attn();
    el[i].ff1_b = F(p++); el[i].ff1_w = F(p++); el[i].ff2_b = F(p++); el[i].ff2_w = F(p++);
    el[i].ln1_b = F(p++); el[i].ln1_g = F(p++); el[i].ln2_b = F(p++); el[i].ln2_g = F(p++);
  }
  const float* enc_norm_b = F(p++);
  const float* enc_norm_g = F(p++);
  const float* proj_b = F(p++);
  const float* proj_w = F(p++);

  /* workspace layout (floats) */
  float* ws = (float*)d_ws;
  size_t off = 0;
  auto take = [&](size_t n) { float* q = ws + off; off += (n + 63) & ~(size_t)63; return q; };
  float* bX  = take(2097152);
  float* bY  = take(2097152);
  float* bT  = take(2097152);
  float* bQ  = take(2097152);
  float* bK  = take(2097152);
  float* bV  = take(2097152);
  float* bH  = take(8388608);
  float* bXP = take(2101248);
  float* bENC = take(524288);
  float* bM  = take(32768);
  float* bS  = take(32 * UMAX * 1024);
  int*   bTop = (int*)take(2048);

  /* ==================== encoder ==================== */
  {
    long tot = 4L * 1024 * DM;
    k_embed<<<(unsigned)((tot + 255) / 256), 256, 0, stream>>>(
        x_enc, xm_enc, enc_conv_w, enc_time_w, enc_time_b, bX, 1024, 7);
  }
  int curL = 1024;
  for (int i = 0; i < 3; ++i) {
    int M = 4 * curL;
    const AttnP& a = el[i].attn;
    gemm(stream, bX, DM, M, 0, a.wq, DM, 1, a.bq, nullptr, bQ, M, DM, DM, 0, 0);
    gemm(stream, bX, DM, M, 0, a.wk, DM, 1, a.bk, nullptr, bK, M, DM, DM, 0, 0);
    gemm(stream, bX, DM, M, 0, a.wv, DM, 1, a.bv, nullptr, bV, M, DM, DM, 0, 0);
    int u = uc_factor(curL);
    run_attn(stream, bQ, bK, bV, bT, bM, bS, bTop, curL, curL, u, u, 0);
    gemm(stream, bT, DM, M, 0, a.wo, DM, 1, a.bo, bX, bY, M, DM, DM, 0, 0);
    k_layernorm<<<M, 256, 0, stream>>>(bY, el[i].ln1_g, el[i].ln1_b, bX);
    gemm(stream, bX, DM, M, 0, el[i].ff1_w, 2048, 1, el[i].ff1_b, nullptr, bH, M, 2048, DM, 1, 0);
    gemm(stream, bH, 2048, M, 0, el[i].ff2_w, DM, 1, el[i].ff2_b, bX, bY, M, DM, 2048, 0, 0);
    k_layernorm<<<M, 256, 0, stream>>>(bY, el[i].ln2_g, el[i].ln2_b, bX);

    if (i < 2) {
      long tot = 4L * (curL + 2) * DM;
      k_padcirc<<<(unsigned)((tot + 255) / 256), 256, 0, stream>>>(bX, bXP, curL);
      for (int t = 0; t < 3; ++t)
        gemm(stream, bXP + t * DM, DM, curL, curL + 2,
             convL_w[i] + t, 3, 1536,                 /* w[o][k][t]: k-stride 3, n-stride 512*3 */
             (t == 2) ? convL_b[i] : nullptr, nullptr,
             bT, M, DM, DM, 0, (t > 0) ? 1 : 0);
      long n = (long)M * DM;
      k_eluscale<<<(unsigned)((n + 255) / 256), 256, 0, stream>>>(bT, n);
      int outL = curL / 2;
      long tot2 = 4L * outL * DM;
      k_maxpool<<<(unsigned)((tot2 + 255) / 256), 256, 0, stream>>>(bT, bX, curL, outL);
      curL = outL;
    }
  }
  k_layernorm<<<4 * curL, 256, 0, stream>>>(bX, enc_norm_g, enc_norm_b, bENC);
  const int LE = curL;  /* 256 */

  /* ==================== decoder ==================== */
  {
    long tot = 4L * 1024 * DM;
    k_embed<<<(unsigned)((tot + 255) / 256), 256, 0, stream>>>(
        x_dec, xm_dec, dec_conv_w, dec_time_w, dec_time_b, bX, 1024, 7);
  }
  const int MD = 4 * 1024;
  const int ME = 4 * LE;
  for (int j = 0; j < 2; ++j) {
    const AttnP& sa = dl[j].self;
    gemm(stream, bX, DM, MD, 0, sa.wq, DM, 1, sa.bq, nullptr, bQ, MD, DM, DM, 0, 0);
    gemm(stream, bX, DM, MD, 0, sa.wk, DM, 1, sa.bk, nullptr, bK, MD, DM, DM, 0, 0);
    gemm(stream, bX, DM, MD, 0, sa.wv, DM, 1, sa.bv, nullptr, bV, MD, DM, DM, 0, 0);
    int uS = uc_factor(1024);
    run_attn(stream, bQ, bK, bV, bT, bM, bS, bTop, 1024, 1024, uS, uS, 1);
    gemm(stream, bT, DM, MD, 0, sa.wo, DM, 1, sa.bo, bX, bY, MD, DM, DM, 0, 0);
    k_layernorm<<<MD, 256, 0, stream>>>(bY, dl[j].ln1_g, dl[j].ln1_b, bX);

    const AttnP& ca = dl[j].cross;
    gemm(stream, bX,   DM, MD, 0, ca.wq, DM, 1, ca.bq, nullptr, bQ, MD, DM, DM, 0, 0);
    gemm(stream, bENC, DM, ME, 0, ca.wk, DM, 1, ca.bk, nullptr, bK, ME, DM, DM, 0, 0);
    gemm(stream, bENC, DM, ME, 0, ca.wv, DM, 1, ca.bv, nullptr, bV, ME, DM, DM, 0, 0);
    run_attn(stream, bQ, bK, bV, bT, bM, bS, bTop, 1024, LE,
             uc_factor(1024), uc_factor(LE), 0);
    gemm(stream, bT, DM, MD, 0, ca.wo, DM, 1, ca.bo, bX, bY, MD, DM, DM, 0, 0);
    k_layernorm<<<MD, 256, 0, stream>>>(bY, dl[j].ln2_g, dl[j].ln2_b, bX);

    gemm(stream, bX, DM, MD, 0, dl[j].ff1_w, 2048, 1, dl[j].ff1_b, nullptr, bH, MD, 2048, DM, 1, 0);
    gemm(stream, bH, 2048, MD, 0, dl[j].ff2_w, DM, 1, dl[j].ff2_b, bX, bY, MD, DM, 2048, 0, 0);
    k_layernorm<<<MD, 256, 0, stream>>>(bY, dl[j].ln3_g, dl[j].ln3_b, bX);
  }
  k_layernorm<<<MD, 256, 0, stream>>>(bX, dec_norm_g, dec_norm_b, bY);
  k_proj<<<(4 * 512 + 127) / 128, 128, 0, stream>>>(bY, proj_w, proj_b, (float*)d_out);
}
